// Encoder_11510512354060
// MI455X (gfx1250) — compile-verified, matrix operand order checked by reading, not run on hardware
//
#include <hip/hip_runtime.h>

// ---------------------------------------------------------------------------
// Roost GNN encoder for MI455X (gfx1250, wave32, WMMA).
// - All Linear layers feeding a BatchNorm run as bf16 WMMA with f32 accum
//   (their biases cancel inside BN and are skipped; mu/logvar keep biases).
// - Edge message GEMM z = [x[src]|x[dst]] @ Wg is done twice (stats pass +
//   apply pass) instead of materializing z (2x409 MB HBM round-trip).
// - Edge-row gathers stage into LDS via GLOBAL_LOAD_ASYNC_TO_LDS_B128 when
//   the builtin is available (ASYNCcnt path), else uint4 copies.
// ---------------------------------------------------------------------------

#define NNODE  50000
#define NEDGE  800000
#define NGRAPH 500
#define HID    128
#define HEADS  4
#define DH     32
#define AIN    200
#define NL     5

typedef __attribute__((ext_vector_type(16))) __bf16 v16bf;
typedef __attribute__((ext_vector_type(8)))  float  v8f;

__device__ __forceinline__ unsigned short f2bf(float f) {
  unsigned u = __builtin_bit_cast(unsigned, f);
  unsigned r = u + 0x7FFFu + ((u >> 16) & 1u);   // round-to-nearest-even
  return (unsigned short)(r >> 16);
}

// ---- async global->LDS staging (gfx1250 ASYNCcnt path), guarded ------------
#if __has_builtin(__builtin_amdgcn_global_load_async_to_lds_b128)
#define HAVE_ASYNC_LDS 1
typedef int v4i_ __attribute__((ext_vector_type(4)));
typedef __attribute__((address_space(1))) v4i_ g_v4i;   // global int4*
typedef __attribute__((address_space(3))) v4i_ l_v4i;   // LDS int4*
__device__ __forceinline__ void async_g2l_b128(const void* g, void* lds) {
  g_v4i* gp = (g_v4i*)(unsigned long long)(uintptr_t)g;
  l_v4i* lp = (l_v4i*)(unsigned)(uintptr_t)lds;         // low 32b = LDS offset
  __builtin_amdgcn_global_load_async_to_lds_b128(gp, lp, 0, 0);
}
__device__ __forceinline__ void wait_async0() {
#if __has_builtin(__builtin_amdgcn_s_wait_asynccnt)
  __builtin_amdgcn_s_wait_asynccnt(0);
#else
  asm volatile("s_wait_asynccnt 0x0" ::: "memory");
#endif
}
#else
#define HAVE_ASYNC_LDS 0
#endif

// bf16 fragment K layout (CDNA5 ISA 7.12.2): lane kh=lane>>4 holds
// K pairs {kh*8+0,2,4,6} then {16+kh*8+0,2,4,6} per 32-deep step -> each half
// is 4 consecutive dwords (16B aligned) => two b128 loads per fragment.
__device__ __forceinline__ int kmap(int v, int kh) {
  return (v < 4) ? (kh * 8 + 2 * v) : (16 + kh * 8 + 2 * (v - 4));
}

// One wave computes a 16(M)x16(N) f32 tile. A: 16 x (NK*32) bf16 strip in LDS
// (row stride 256 elems = 512B). B: pre-packed fragments, 8 dwords per
// (tile,kstep,lane), contiguous.
template <int NK>
__device__ __forceinline__ v8f tile_gemm(const unsigned short* As,
                                         const unsigned* Bpack,
                                         int wave, int lane) {
  v8f c = {0.f, 0.f, 0.f, 0.f, 0.f, 0.f, 0.f, 0.f};
  const int r  = lane & 15;
  const int kh = lane >> 4;
  const uint4* A4 = (const uint4*)As;                 // 16B granules
  const uint4* B4 = (const uint4*)Bpack;
  const int abase = r * 32 + kh;                      // + ks*4 (+2 for hi half)
  const int bbase = (wave * NK * 32 + lane) * 2;      // + ks*64
#pragma unroll
  for (int ks = 0; ks < NK; ++ks) {
    union { v16bf v; uint4 q[2]; } a, b;
    a.q[0] = A4[abase + ks * 4];                      // ds_load_b128
    a.q[1] = A4[abase + ks * 4 + 2];                  // ds_load_b128
    b.q[0] = B4[bbase + ks * 64];                     // global_load_b128
    b.q[1] = B4[bbase + ks * 64 + 1];
    c = __builtin_amdgcn_wmma_f32_16x16x32_bf16(false, a.v, false, b.v,
                                                (short)0, c, false, false);
  }
  return c;
}

// ---- weight packing into WMMA B-fragment order ----------------------------
// layout: [ntile(8)][kstep(nk)][lane(32)][v(8)] dwords, each dword = bf16 pair.
// mode 0: B[k*128 + c] (row-major KxN). mode 1: Wg [HEADS][256][DH], c=h*32+d.
__global__ void k_packB(const float* __restrict__ B, int Kreal, int nk,
                        int mode, unsigned* __restrict__ Bpack) {
  int total = 8 * nk * 32 * 8;
  for (int id = blockIdx.x * blockDim.x + threadIdx.x; id < total;
       id += gridDim.x * blockDim.x) {
    int v = id & 7, lane = (id >> 3) & 31, rem = id >> 8;
    int ks = rem % nk, t = rem / nk;
    int c = (t << 4) + (lane & 15), kh = lane >> 4;
    int k0 = (ks << 5) + kmap(v, kh);
    float lo = 0.f, hi = 0.f;
    if (k0 < Kreal)
      lo = (mode == 0) ? B[k0 * HID + c]
                       : B[(c >> 5) * (256 * DH) + k0 * DH + (c & 31)];
    if (k0 + 1 < Kreal)
      hi = (mode == 0) ? B[(k0 + 1) * HID + c]
                       : B[(c >> 5) * (256 * DH) + (k0 + 1) * DH + (c & 31)];
    Bpack[id] = (unsigned)f2bf(lo) | ((unsigned)f2bf(hi) << 16);
  }
}

// ---- generic row GEMM: out[R x 128] = A[R x Kreal] @ B (+bias) -------------
template <int NK>
__global__ __launch_bounds__(256)
void k_gemm(const float* __restrict__ A, int R, int lda, int Kreal,
            const unsigned* __restrict__ Bpack, const float* __restrict__ bias,
            float* __restrict__ out) {
  constexpr int KPAD = NK * 32;
  __shared__ __align__(16) unsigned short As[16 * 256];
  const int lane = threadIdx.x & 31, wave = threadIdx.x >> 5;
  const int ntiles = (R + 15) >> 4;
  unsigned* Asd = (unsigned*)As;
  for (int tile = blockIdx.x; tile < ntiles; tile += gridDim.x) {
    int row0 = tile << 4;
    for (int idx = threadIdx.x; idx < 16 * (KPAD / 2); idx += blockDim.x) {
      int r = idx / (KPAD / 2), kd = idx - r * (KPAD / 2);
      int k = kd * 2, row = row0 + r;
      float v0 = 0.f, v1 = 0.f;
      if (row < R) {
        const float* Ar = A + (long long)row * lda;
        if (k < Kreal) v0 = Ar[k];
        if (k + 1 < Kreal) v1 = Ar[k + 1];
      }
      Asd[(r << 7) + kd] = (unsigned)f2bf(v0) | ((unsigned)f2bf(v1) << 16);
    }
    __syncthreads();
    v8f c = tile_gemm<NK>(As, Bpack, wave, lane);
    int col = (wave << 4) + (lane & 15);
    int rbase = (lane >> 4) * 8;
    float bv = bias ? bias[col] : 0.f;
#pragma unroll
    for (int v = 0; v < 8; ++v) {
      int row = row0 + rbase + v;
      if (row < R) out[(long long)row * HID + col] = c[v] + bv;
    }
    __syncthreads();
  }
}

// stage 16 gathered concat rows [x[src]|x[dst]] (bf16) into LDS
__device__ __forceinline__ void stage_edges(unsigned short* As,
                                            const unsigned short* xb,
                                            const int* sS, const int* dS) {
  uint4* As4 = (uint4*)As;
  const uint4* xb4 = (const uint4*)xb;                // 16 uint4 per node row
  for (int idx = threadIdx.x; idx < 16 * 32; idx += blockDim.x) {
    int r = idx >> 5, q = idx & 31;                   // 32 uint4 per concat row
    int node = (q < 16) ? sS[r] : dS[r];
    int o = (q < 16) ? q : q - 16;
#if HAVE_ASYNC_LDS
    async_g2l_b128(&xb4[(long long)node * 16 + o], &As4[(r << 5) + q]);
#else
    As4[(r << 5) + q] = xb4[(long long)node * 16 + o];
#endif
  }
#if HAVE_ASYNC_LDS
  wait_async0();
#endif
}

// ---- roost message GEMM, pass 1: per-channel stats of z (no store) ---------
__global__ __launch_bounds__(256)
void k_zgemm_stats(const unsigned short* __restrict__ xb,
                   const int* __restrict__ src, const int* __restrict__ dst,
                   const unsigned* __restrict__ Bpack,
                   float* __restrict__ zsum, float* __restrict__ zsq) {
  __shared__ __align__(16) unsigned short As[16 * 256];
  __shared__ int sS[16], dS[16];
  __shared__ float lsum[HID], lsq[HID];
  const int lane = threadIdx.x & 31, wave = threadIdx.x >> 5;
  if (threadIdx.x < HID) { lsum[threadIdx.x] = 0.f; lsq[threadIdx.x] = 0.f; }
  const int ntiles = NEDGE >> 4;
  for (int tile = blockIdx.x; tile < ntiles; tile += gridDim.x) {
    int e0 = tile << 4;
    if (threadIdx.x < 16) {
      sS[threadIdx.x] = src[e0 + threadIdx.x];
      dS[threadIdx.x] = dst[e0 + threadIdx.x];
    }
    __syncthreads();
    stage_edges(As, xb, sS, dS);
    __syncthreads();
    v8f c = tile_gemm<8>(As, Bpack, wave, lane);
    int col = (wave << 4) + (lane & 15);
    float s = 0.f, q = 0.f;
#pragma unroll
    for (int v = 0; v < 8; ++v) { s += c[v]; q += c[v] * c[v]; }
    atomicAdd(&lsum[col], s);
    atomicAdd(&lsq[col], q);
    __syncthreads();
  }
  if (threadIdx.x < HID) {
    atomicAdd(&zsum[threadIdx.x], lsum[threadIdx.x]);
    atomicAdd(&zsq[threadIdx.x], lsq[threadIdx.x]);
  }
}

// ---- roost message GEMM, pass 2: recompute z, BN+SiLU, weight, scatter -----
__global__ __launch_bounds__(256)
void k_zgemm_apply(const unsigned short* __restrict__ xb,
                   const int* __restrict__ src, const int* __restrict__ dst,
                   const unsigned* __restrict__ Bpack,
                   const float* __restrict__ zscale, const float* __restrict__ zshift,
                   const float* __restrict__ wexp, const float* __restrict__ sumw,
                   float* __restrict__ h) {
  __shared__ __align__(16) unsigned short As[16 * 256];
  __shared__ int sS[16], dS[16];
  __shared__ float aS[16];
  const int lane = threadIdx.x & 31, wave = threadIdx.x >> 5;
  const int ntiles = NEDGE >> 4;
  for (int tile = blockIdx.x; tile < ntiles; tile += gridDim.x) {
    int e0 = tile << 4;
    if (threadIdx.x < 16) {
      int e = e0 + threadIdx.x;
      sS[threadIdx.x] = src[e];
      int d = dst[e];
      dS[threadIdx.x] = d;
      aS[threadIdx.x] = wexp[e] / sumw[d];            // attention coefficient
    }
    __syncthreads();
    stage_edges(As, xb, sS, dS);
    __syncthreads();
    v8f c = tile_gemm<8>(As, Bpack, wave, lane);
    int col = (wave << 4) + (lane & 15);
    int rbase = (lane >> 4) * 8;
    float sc = zscale[col], sh = zshift[col];
#pragma unroll
    for (int v = 0; v < 8; ++v) {
      int r = rbase + v;
      float zv = sc * c[v] + sh;
      float zb = zv / (1.f + __expf(-zv));            // SiLU
      atomicAdd(&h[(long long)dS[r] * HID + col], aS[r] * zb);
    }
    __syncthreads();
  }
}

// ---- per-edge attention scalar path ---------------------------------------
__global__ void k_edge_f(const float* __restrict__ x, const float* __restrict__ Wf,
                         const int* __restrict__ src, const int* __restrict__ dst,
                         float* __restrict__ f) {
  __shared__ float wf[2 * HID];
  for (int i = threadIdx.x; i < 2 * HID; i += blockDim.x) wf[i] = Wf[i];
  __syncthreads();
  int stride = gridDim.x * blockDim.x;
  for (int e = blockIdx.x * blockDim.x + threadIdx.x; e < NEDGE; e += stride) {
    const float4* xs = (const float4*)(x + (long long)src[e] * HID);
    const float4* xd = (const float4*)(x + (long long)dst[e] * HID);
    float acc = 0.f;
#pragma unroll 8
    for (int k = 0; k < HID / 4; ++k) {
      float4 a = xs[k], b = xd[k];
      acc += a.x * wf[4 * k] + a.y * wf[4 * k + 1] + a.z * wf[4 * k + 2] + a.w * wf[4 * k + 3];
      acc += b.x * wf[HID + 4 * k] + b.y * wf[HID + 4 * k + 1] +
             b.z * wf[HID + 4 * k + 2] + b.w * wf[HID + 4 * k + 3];
    }
    f[e] = acc;
  }
}

__global__ void k_edge_att(const float* __restrict__ f, const float* __restrict__ w,
                           const int* __restrict__ src, const int* __restrict__ dst,
                           const float* __restrict__ fscale, const float* __restrict__ fshift,
                           float* __restrict__ wexp, float* __restrict__ sumw) {
  float sc = fscale[0], sh = fshift[0];
  int stride = gridDim.x * blockDim.x;
  for (int e = blockIdx.x * blockDim.x + threadIdx.x; e < NEDGE; e += stride) {
    float t = sc * f[e] + sh;
    float s = t / (1.f + __expf(-t));                 // SiLU
    float wv = w[src[e]] * __expf(s);
    wexp[e] = wv;
    atomicAdd(&sumw[dst[e]], wv);
  }
}

// ---- column stats / BN helpers --------------------------------------------
__global__ void k_colstats(const float* __restrict__ A, long long total, int C,
                           float* __restrict__ sum, float* __restrict__ sq) {
  // blockDim (256) is a multiple of C (1 or 128) -> each thread owns one col.
  __shared__ float ls[HID], lq[HID];
  if (threadIdx.x < C) { ls[threadIdx.x] = 0.f; lq[threadIdx.x] = 0.f; }
  __syncthreads();
  int c = (C == 1) ? 0 : (threadIdx.x & (C - 1));
  float s = 0.f, q = 0.f;
  long long stride = (long long)gridDim.x * blockDim.x;
  for (long long i = (long long)blockIdx.x * blockDim.x + threadIdx.x; i < total; i += stride) {
    float v = A[i];
    s += v; q += v * v;
  }
  atomicAdd(&ls[c], s);
  atomicAdd(&lq[c], q);
  __syncthreads();
  if (threadIdx.x < C) {
    atomicAdd(&sum[threadIdx.x], ls[threadIdx.x]);
    atomicAdd(&sq[threadIdx.x], lq[threadIdx.x]);
  }
}

__global__ void k_finalize(const float* __restrict__ sum, const float* __restrict__ sq,
                           float invR, const float* __restrict__ g, const float* __restrict__ be,
                           float* __restrict__ scale, float* __restrict__ shift, int C) {
  int c = threadIdx.x;
  if (c < C) {
    float m = sum[c] * invR;
    float v = sq[c] * invR - m * m;
    float sc = g[c] * rsqrtf(v + 1e-5f);
    scale[c] = sc;
    shift[c] = be[c] - m * sc;
  }
}

// ---- elementwise helpers ---------------------------------------------------
__global__ void k_zero(float* __restrict__ p, long long n) {
  long long stride = (long long)gridDim.x * blockDim.x;
  for (long long i = (long long)blockIdx.x * blockDim.x + threadIdx.x; i < n; i += stride)
    p[i] = 0.f;
}

__global__ void k_bn_silu(const float* __restrict__ y, float* __restrict__ x,
                          const float* __restrict__ scale, const float* __restrict__ shift,
                          long long n) {
  long long stride = (long long)gridDim.x * blockDim.x;
  for (long long i = (long long)blockIdx.x * blockDim.x + threadIdx.x; i < n; i += stride) {
    int c = (int)(i & (HID - 1));
    float t = scale[c] * y[i] + shift[c];
    x[i] = t / (1.f + __expf(-t));
  }
}

__global__ void k_bn_res(const float* __restrict__ h, float* __restrict__ x,
                         const float* __restrict__ scale, const float* __restrict__ shift,
                         long long n) {
  long long stride = (long long)gridDim.x * blockDim.x;
  for (long long i = (long long)blockIdx.x * blockDim.x + threadIdx.x; i < n; i += stride) {
    int c = (int)(i & (HID - 1));
    x[i] += scale[c] * h[i] + shift[c];
  }
}

__global__ void k_cast(const float* __restrict__ x, unsigned short* __restrict__ xb, long long n) {
  long long stride = (long long)gridDim.x * blockDim.x;
  for (long long i = (long long)blockIdx.x * blockDim.x + threadIdx.x; i < n; i += stride)
    xb[i] = f2bf(x[i]);
}

__global__ void k_reparam(float* __restrict__ x, const float* __restrict__ eps,
                          const float* __restrict__ mu, const float* __restrict__ lv,
                          float* __restrict__ kld) {
  float acc = 0.f;
  long long n = (long long)NNODE * HID;
  long long stride = (long long)gridDim.x * blockDim.x;
  for (long long i = (long long)blockIdx.x * blockDim.x + threadIdx.x; i < n; i += stride) {
    float l = lv[i], m = mu[i];
    float el = __expf(l);
    x[i] = eps[i] * __expf(0.5f * l) + m;
    acc += -0.5f * (1.f + l - m * m - el);
  }
  __shared__ float red[256];
  red[threadIdx.x] = acc;
  __syncthreads();
  for (int s = blockDim.x >> 1; s; s >>= 1) {
    if (threadIdx.x < s) red[threadIdx.x] += red[threadIdx.x + s];
    __syncthreads();
  }
  if (threadIdx.x == 0) atomicAdd(kld, red[0]);
}

__global__ void k_pool(const float* __restrict__ x, const int* __restrict__ gid,
                       float* __restrict__ sums, float* __restrict__ cnt) {
  long long n = (long long)NNODE * HID;
  long long stride = (long long)gridDim.x * blockDim.x;
  for (long long i = (long long)blockIdx.x * blockDim.x + threadIdx.x; i < n; i += stride) {
    int node = (int)(i >> 7), c = (int)(i & (HID - 1));
    atomicAdd(&sums[(long long)gid[node] * HID + c], x[i]);
  }
  for (long long i = (long long)blockIdx.x * blockDim.x + threadIdx.x; i < NNODE; i += stride)
    atomicAdd(&cnt[gid[(int)i]], 1.f);
}

__global__ void k_pool_out(const float* __restrict__ sums, const float* __restrict__ cnt,
                           const float* __restrict__ kld, float* __restrict__ out) {
  int i = blockIdx.x * blockDim.x + threadIdx.x;
  if (i < NGRAPH * HID) {
    int g = i >> 7;
    out[i] = sums[i] / fmaxf(cnt[g], 1.f);
  }
  if (i == 0) out[NGRAPH * HID] = kld[0] / (float)NNODE;
}

// ---------------------------------------------------------------------------
extern "C" void kernel_launch(void* const* d_in, const int* in_sizes, int n_in,
                              void* d_out, int out_size, void* d_ws, size_t ws_size,
                              hipStream_t stream) {
  (void)in_sizes; (void)n_in; (void)out_size; (void)ws_size;
  const float* atom   = (const float*)d_in[0];
  const float* nodew  = (const float*)d_in[1];
  const float* eps    = (const float*)d_in[2];
  const int*   src    = (const int*)d_in[3];
  const int*   dst    = (const int*)d_in[4];
  const int*   gid    = (const int*)d_in[5];
  const float* W_embed = (const float*)d_in[6];
  // d_in[7] b_embed: cancels inside BatchNorm
  const float* g_embed  = (const float*)d_in[8];
  const float* be_embed = (const float*)d_in[9];
  const float* Wf  = (const float*)d_in[10];
  // d_in[11] bf: cancels inside BatchNorm
  const float* gf  = (const float*)d_in[12];
  const float* bef = (const float*)d_in[13];
  const float* Wg  = (const float*)d_in[14];
  // d_in[15] bg: cancels inside BatchNorm
  const float* gg  = (const float*)d_in[16];
  const float* beg = (const float*)d_in[17];
  const float* gn  = (const float*)d_in[18];
  const float* ben = (const float*)d_in[19];
  const float* W_mu  = (const float*)d_in[20];
  const float* b_mu  = (const float*)d_in[21];
  const float* W_var = (const float*)d_in[22];
  const float* b_var = (const float*)d_in[23];
  float* out = (float*)d_out;

  // ---- workspace carve-up ----
  char* ws = (char*)d_ws;
  size_t off = 0;
  auto alloc = [&](size_t bytes) -> char* {
    off = (off + 255) & ~(size_t)255;
    char* p = ws + off;
    off += bytes;
    return p;
  };
  float*          x      = (float*)alloc((size_t)NNODE * HID * 4);
  unsigned short* xb     = (unsigned short*)alloc((size_t)NNODE * HID * 2);
  float*          h      = (float*)alloc((size_t)NNODE * HID * 4);   // also embed pre-BN
  float*          mu     = (float*)alloc((size_t)NNODE * HID * 4);
  float*          logv   = (float*)alloc((size_t)NNODE * HID * 4);
  float*          fbuf   = (float*)alloc((size_t)NEDGE * 4);
  float*          wexp   = (float*)alloc((size_t)NEDGE * 4);
  float*          sumw   = (float*)alloc((size_t)NNODE * 4);
  unsigned*       BpE    = (unsigned*)alloc((size_t)8 * 7 * 32 * 8 * 4);
  unsigned*       BpM    = (unsigned*)alloc((size_t)8 * 4 * 32 * 8 * 4);
  unsigned*       BpV    = (unsigned*)alloc((size_t)8 * 4 * 32 * 8 * 4);
  unsigned*       BpG    = (unsigned*)alloc((size_t)NL * 8 * 8 * 32 * 8 * 4);
  float*          stat   = (float*)alloc(512 * 4);   // sum|sq|scale|shift x128
  float*          psum   = (float*)alloc((size_t)NGRAPH * HID * 4);
  float*          pcnt   = (float*)alloc((size_t)NGRAPH * 4);
  float*          kld    = (float*)alloc(4);

  float* ssum = stat, * ssq = stat + 128, * sscale = stat + 256, * sshift = stat + 384;

  const int T = 256;
  const int GB = 2048;   // generic GEMM blocks (grid-stride over row tiles)
  const int ZB = 4096;   // edge GEMM blocks
  const int EW = 2048;   // elementwise blocks
  const long long NH = (long long)NNODE * HID;

  auto zero = [&](float* p, long long n) { k_zero<<<1024, T, 0, stream>>>(p, n); };

  // ---- pack weights into WMMA B-fragment layout ----
  k_packB<<<64, T, 0, stream>>>(W_embed, AIN, 7, 0, BpE);
  k_packB<<<64, T, 0, stream>>>(W_mu,  HID, 4, 0, BpM);
  k_packB<<<64, T, 0, stream>>>(W_var, HID, 4, 0, BpV);
  for (int l = 0; l < NL; ++l)
    k_packB<<<64, T, 0, stream>>>(Wg + (size_t)l * HEADS * 256 * DH, 2 * HID, 8, 1,
                                  BpG + (size_t)l * 8 * 8 * 32 * 8);

  // ---- atom embedding: GEMM -> BN -> SiLU ----
  k_gemm<7><<<GB, T, 0, stream>>>(atom, NNODE, AIN, AIN, BpE, nullptr, h);
  zero(stat, 256);
  k_colstats<<<EW, T, 0, stream>>>(h, NH, HID, ssum, ssq);
  k_finalize<<<1, 128, 0, stream>>>(ssum, ssq, 1.f / NNODE, g_embed, be_embed, sscale, sshift, HID);
  k_bn_silu<<<EW, T, 0, stream>>>(h, x, sscale, sshift, NH);
  k_cast<<<EW, T, 0, stream>>>(x, xb, NH);

  auto roost = [&](int l) {
    const unsigned* Bg = BpG + (size_t)l * 8 * 8 * 32 * 8;
    // attention gate f = hihj @ Wf; BN over E (1 channel)
    k_edge_f<<<EW, T, 0, stream>>>(x, Wf + (size_t)l * 2 * HID, src, dst, fbuf);
    zero(stat, 256);
    k_colstats<<<EW, T, 0, stream>>>(fbuf, NEDGE, 1, ssum, ssq);
    k_finalize<<<1, 128, 0, stream>>>(ssum, ssq, 1.f / NEDGE, gf + l, bef + l, sscale, sshift, 1);
    // e = exp(silu(bn(f))); wexp = w[src]*e; sum_w = segsum_dst(wexp)
    zero(sumw, NNODE);
    k_edge_att<<<EW, T, 0, stream>>>(fbuf, nodew, src, dst, sscale, sshift, wexp, sumw);
    // z stats pass (WMMA, no store), then finalize per-channel BN
    zero(stat, 256);
    k_zgemm_stats<<<ZB, T, 0, stream>>>(xb, src, dst, Bg, ssum, ssq);
    k_finalize<<<1, 128, 0, stream>>>(ssum, ssq, 1.f / NEDGE, gg + (size_t)l * HID,
                                      beg + (size_t)l * HID, sscale, sshift, HID);
    // z apply pass (WMMA recompute) + weighted scatter into h
    zero(h, NH);
    k_zgemm_apply<<<ZB, T, 0, stream>>>(xb, src, dst, Bg, sscale, sshift, wexp, sumw, h);
    // node BN + residual
    zero(stat, 256);
    k_colstats<<<EW, T, 0, stream>>>(h, NH, HID, ssum, ssq);
    k_finalize<<<1, 128, 0, stream>>>(ssum, ssq, 1.f / NNODE, gn + (size_t)l * HID,
                                      ben + (size_t)l * HID, sscale, sshift, HID);
    k_bn_res<<<EW, T, 0, stream>>>(h, x, sscale, sshift, NH);
    k_cast<<<EW, T, 0, stream>>>(x, xb, NH);
  };

  for (int l = 0; l < NL; ++l) {
    roost(l);
    if (l == 0) {  // reparameterization after the NBASE(=1) base layer
      k_gemm<4><<<GB, T, 0, stream>>>(x, NNODE, HID, HID, BpM, b_mu, mu);
      k_gemm<4><<<GB, T, 0, stream>>>(x, NNODE, HID, HID, BpV, b_var, logv);
      zero(kld, 1);
      k_reparam<<<EW, T, 0, stream>>>(x, eps, mu, logv, kld);
      k_cast<<<EW, T, 0, stream>>>(x, xb, NH);
    }
  }

  // ---- graph average pooling + KLD ----
  zero(psum, (long long)NGRAPH * HID);
  zero(pcnt, NGRAPH);
  k_pool<<<EW, T, 0, stream>>>(x, gid, psum, pcnt);
  k_pool_out<<<(NGRAPH * HID + T - 1) / T, T, 0, stream>>>(psum, pcnt, kld, out);
}
